// AttractorGuidedNeuroFocalAttention_41051297415243
// MI455X (gfx1250) — matrix operand order repaired and checked by reading, not compiled, not run on hardware
//
#include <hip/hip_runtime.h>
#include <hip/hip_bf16.h>
#include <math.h>

typedef __attribute__((ext_vector_type(16))) _Float16 v16h;
typedef __attribute__((ext_vector_type(8)))  _Float16 v8h;
typedef __attribute__((ext_vector_type(8)))  float    v8f;

#define WMMA_F16(A, B, C) \
    __builtin_amdgcn_wmma_f32_16x16x32_f16(false, (A), false, (B), (short)0, (C), false, false)

// Problem constants
#define BB    2
#define NN    2048
#define DIM   512
#define HH    8
#define DH    64
#define MTOT  (BB * NN)          // 4096 rows
#define INV_SQRT_DH 0.125f       // 1/sqrt(64)

// Build a v16h fragment from two 16-byte contiguous chunks.
__device__ __forceinline__ v16h ld2x8(const _Float16* p0, const _Float16* p1) {
    v8h lo = *(const v8h*)p0;
    v8h hi = *(const v8h*)p1;
    return __builtin_shufflevector(lo, hi, 0, 1, 2, 3, 4, 5, 6, 7,
                                           8, 9, 10, 11, 12, 13, 14, 15);
}

// LDS byte offset of a generic pointer into __shared__ (ISA 10.2: low 32 bits
// of an LDS-aperture flat address are the LDS offset).
__device__ __forceinline__ unsigned lds_off(const void* p) {
    return (unsigned)(unsigned long long)p;
}

// Async DMA: 16 bytes per lane, global -> LDS, tracked by ASYNCcnt.
__device__ __forceinline__ void async_copy_b128(unsigned lds_dst,
                                                const _Float16* gsrc) {
    asm volatile("global_load_async_to_lds_b128 %0, %1, off"
                 :: "v"(lds_dst), "v"(gsrc)
                 : "memory");
}

// ---------------------------------------------------------------------------
// fp32 -> f16 conversion (elementwise), used for z
// ---------------------------------------------------------------------------
__global__ void f32_to_f16_kernel(const float* __restrict__ in,
                                  _Float16* __restrict__ out, int n) {
    int i = blockIdx.x * 256 + threadIdx.x;
    if (i < n) out[i] = (_Float16)in[i];
}

// ---------------------------------------------------------------------------
// fp32 [512,512] (k-major) -> f16 transposed [n][k], LDS-tiled for coalescing
// ---------------------------------------------------------------------------
__global__ void transpose_w_kernel(const float* __restrict__ in,
                                   _Float16* __restrict__ out) {
    __shared__ float tile[32][33];
    const int k0 = blockIdx.x * 32, n0 = blockIdx.y * 32;
    const int tx = threadIdx.x & 31, ty = threadIdx.x >> 5;   // 32 x 8
#pragma unroll
    for (int r = ty; r < 32; r += 8)
        tile[r][tx] = in[(size_t)(k0 + r) * DIM + n0 + tx];
    __syncthreads();
#pragma unroll
    for (int r = ty; r < 32; r += 8)
        out[(size_t)(n0 + r) * DIM + k0 + tx] = (_Float16)tile[tx][r];
}

// ---------------------------------------------------------------------------
// Fused Q/K/V projection: z[4096,512] x W^T + b (3 weights, shared A-frags)
// Q,K stored head-major f16: Qh[(bh*N + n)*64 + d]
// V stored head-transposed:  Vt[(bh*64 + d)*N + n]
// ---------------------------------------------------------------------------
__global__ void gemm_qkv_kernel(const _Float16* __restrict__ zh,
                                const _Float16* __restrict__ Wqt,
                                const _Float16* __restrict__ Wkt,
                                const _Float16* __restrict__ Wvt,
                                const float* __restrict__ bq,
                                const float* __restrict__ bk,
                                const float* __restrict__ bv,
                                _Float16* __restrict__ Qh,
                                _Float16* __restrict__ Kh,
                                _Float16* __restrict__ Vt) {
    const int lane = threadIdx.x & 31;
    const int wave = threadIdx.x >> 5;
    const int row0 = blockIdx.x * 16;                // M tile
    const int col0 = blockIdx.y * 128 + wave * 16;   // N tile (per wave)

    const int rowA = row0 + (lane & 15);
    const int nB   = col0 + (lane & 15);
    const int kbA  = (lane >> 4) * 8;
    const int koB  = (lane >> 4) * 16;

    const _Float16* aRow  = zh  + (size_t)rowA * DIM;
    const _Float16* bqRow = Wqt + (size_t)nB * DIM;
    const _Float16* bkRow = Wkt + (size_t)nB * DIM;
    const _Float16* bvRow = Wvt + (size_t)nB * DIM;

    v8f cq = {}, ck = {}, cv = {};

    for (int k0 = 0; k0 < DIM; k0 += 32) {
        const v16h a = ld2x8(aRow + k0 + kbA, aRow + k0 + kbA + 16);

        if (k0 + 32 < DIM) {
            __builtin_prefetch(bqRow + k0 + 32 + koB, 0, 1);
            __builtin_prefetch(bkRow + k0 + 32 + koB, 0, 1);
            __builtin_prefetch(bvRow + k0 + 32 + koB, 0, 1);
        }

        v16h b;
        b  = ld2x8(bqRow + k0 + kbA, bqRow + k0 + kbA + 16);
        cq = WMMA_F16(a, b, cq);
        b  = ld2x8(bkRow + k0 + kbA, bkRow + k0 + kbA + 16);
        ck = WMMA_F16(a, b, ck);
        b  = ld2x8(bvRow + k0 + kbA, bvRow + k0 + kbA + 16);
        cv = WMMA_F16(a, b, cv);
    }

    const float biasQ = bq[nB], biasK = bk[nB], biasV = bv[nB];
    const int h  = col0 >> 6;
    const int d0 = (col0 & 63) + (lane & 15);

#pragma unroll
    for (int r = 0; r < 8; ++r) {
        const int m   = ((lane >> 4) << 3) + r;
        const int row = row0 + m;
        const int b_  = row >> 11;       // / N
        const int n   = row & (NN - 1);
        const int bh  = b_ * HH + h;
        const size_t dqk = ((size_t)bh * NN + n) * DH + d0;
        Qh[dqk] = (_Float16)(cq[r] + biasQ);
        Kh[dqk] = (_Float16)(ck[r] + biasK);
        Vt[((size_t)bh * DH + d0) * NN + n] = (_Float16)(cv[r] + biasV);
    }
}

// ---------------------------------------------------------------------------
// Per-row stats over DH=64: q2 = sum(Q^2), k2 = sum(K^2),
// wf = 1/(exp(-nf)+1) with nf = sum|Q-mu| / (sigma+eps)
// ---------------------------------------------------------------------------
__global__ void row_stats_kernel(const _Float16* __restrict__ Qh,
                                 const _Float16* __restrict__ Kh,
                                 float* __restrict__ q2,
                                 float* __restrict__ k2,
                                 float* __restrict__ wf) {
    const int lane = threadIdx.x & 31;
    const int wave = threadIdx.x >> 5;
    const int row  = blockIdx.x * 8 + wave;   // 0 .. B*H*N-1

    const _Float16* q = Qh + (size_t)row * DH;
    float x0 = (float)q[lane], x1 = (float)q[lane + 32];
    float s  = x0 + x1;
    float ss = x0 * x0 + x1 * x1;
#pragma unroll
    for (int m = 16; m; m >>= 1) {
        s  += __shfl_xor(s,  m, 32);
        ss += __shfl_xor(ss, m, 32);
    }
    const float mu    = s * (1.0f / 64.0f);
    const float var   = ss * (1.0f / 64.0f) - mu * mu;
    const float sigma = sqrtf(var + 1e-5f);
    float ad = fabsf(x0 - mu) + fabsf(x1 - mu);
#pragma unroll
    for (int m = 16; m; m >>= 1) ad += __shfl_xor(ad, m, 32);
    const float nf = ad / (sigma + 1e-5f);

    const _Float16* kk = Kh + (size_t)row * DH;
    float y0 = (float)kk[lane], y1 = (float)kk[lane + 32];
    float ks = y0 * y0 + y1 * y1;
#pragma unroll
    for (int m = 16; m; m >>= 1) ks += __shfl_xor(ks, m, 32);

    if (lane == 0) {
        q2[row] = ss;
        k2[row] = ks;
        wf[row] = 1.0f / (__expf(-nf) + 1.0f);   // DELTA = 1
    }
}

// ---------------------------------------------------------------------------
// Flash-style attention. All 8 waves of a block share one (b,h); K/V tiles of
// 32 keys are double-buffered in LDS via async global->LDS DMA (ASYNCcnt),
// overlapping the next tile's DMA with compute on the current one.
// logits = qk/8 - (q2 + k2 - 2 qk) * wf     (ALPHA = DELTA = 1)
// ---------------------------------------------------------------------------
__global__ void attention_kernel(const _Float16* __restrict__ Qh,
                                 const _Float16* __restrict__ Kh,
                                 const _Float16* __restrict__ Vt,
                                 const float* __restrict__ q2g,
                                 const float* __restrict__ k2g,
                                 const float* __restrict__ wfg,
                                 _Float16* __restrict__ Oh) {
    __shared__ __align__(16) _Float16 Ktile[2][32 * DH];   // [buf][key][d]
    __shared__ __align__(16) _Float16 Vtile[2][DH * 32];   // [buf][d][key]
    __shared__ __align__(16) _Float16 Plds[8][16][32];     // per-wave P

    const int tid  = threadIdx.x;
    const int lane = tid & 31;
    const int wave = tid >> 5;
    const int qt   = blockIdx.x * 8 + wave;        // query tile id, 0..2047
    const int bh   = qt >> 7;                      // (b*H + h), block-uniform
    const int q0   = (qt & 127) << 4;              // first query row in head

    const _Float16* Qp = Qh + ((size_t)bh * NN + q0) * DH;
    const _Float16* Kp = Kh + (size_t)bh * NN * DH;
    const _Float16* Vp = Vt + (size_t)bh * DH * NN;   // [d][key]
    const float* q2p = q2g + bh * NN + q0;
    const float* wfp = wfg + bh * NN + q0;
    const float* k2p = k2g + bh * NN;

    const int mrow = lane & 15;
    const int kbA  = (lane >> 4) * 8;
    const int koB  = (lane >> 4) * 16;

    // Per-thread DMA source/dest (16 bytes each; K tile contiguous 4KB,
    // V tile is 64 rows of 64B with global stride NN halves).
    const _Float16* kSrc = Kp + tid * 8;                        // + j0*DH
    const _Float16* vSrc = Vp + (size_t)(tid >> 2) * NN + (tid & 3) * 8; // + j0
    const unsigned kDst[2] = { lds_off(&Ktile[0][0]) + (unsigned)tid * 16,
                               lds_off(&Ktile[1][0]) + (unsigned)tid * 16 };
    const unsigned vDst[2] = { lds_off(&Vtile[0][0]) + (unsigned)tid * 16,
                               lds_off(&Vtile[1][0]) + (unsigned)tid * 16 };

    // Q A-fragments (d = 0..31 and 32..63), held for the whole key loop
    const _Float16* qBase = Qp + mrow * DH;
    const v16h aq0 = ld2x8(qBase + kbA,      qBase + kbA + 16);
    const v16h aq1 = ld2x8(qBase + 32 + kbA, qBase + 32 + kbA + 16);

    float q2r[8], wfr[8], rmax[8], rsum[8];
#pragma unroll
    for (int r = 0; r < 8; ++r) {
        const int m = ((lane >> 4) << 3) + r;
        q2r[r] = q2p[m];
        wfr[r] = wfp[m];
        rmax[r] = -3.0e38f;
        rsum[r] = 0.0f;
    }

    v8f acc[4] = {{}, {}, {}, {}};   // 16 x 64 output accumulator

    // Prologue: DMA tile 0 into buffer 0
    async_copy_b128(kDst[0], kSrc);
    async_copy_b128(vDst[0], vSrc);

    for (int j0 = 0; j0 < NN; j0 += 32) {
        const int buf = (j0 >> 5) & 1;

        __syncthreads();   // everyone done reading buffer buf^1 (prev tile)

        if (j0 + 32 < NN) {
            // DMA next tile into the other buffer, overlap with compute
            async_copy_b128(kDst[buf ^ 1], kSrc + (j0 + 32) * DH);
            async_copy_b128(vDst[buf ^ 1], vSrc + (j0 + 32));
            asm volatile("s_wait_asynccnt 0x2" ::: "memory");
        } else {
            asm volatile("s_wait_asynccnt 0x0" ::: "memory");
        }
        __syncthreads();   // current buffer complete for all waves

        const _Float16* Kt  = &Ktile[buf][0];
        const _Float16* Vtl = &Vtile[buf][0];

        // ---- S = Q K^T for two 16-key tiles, from LDS ----
        v8f c0 = {}, c1 = {};
        {
            const _Float16* k0p = Kt + (lane & 15) * DH + kbA;
            const _Float16* k1p = Kt + (16 + (lane & 15)) * DH + kbA;
            v16h b;
            b  = ld2x8(k0p, k0p + 16);
            c0 = WMMA_F16(aq0, b, c0);
            b  = ld2x8(k0p + 32, k0p + 48);
            c0 = WMMA_F16(aq1, b, c0);
            b  = ld2x8(k1p, k1p + 16);
            c1 = WMMA_F16(aq0, b, c1);
            b  = ld2x8(k1p + 32, k1p + 48);
            c1 = WMMA_F16(aq1, b, c1);
        }

        const float k2a = k2p[j0 + (lane & 15)];
        const float k2b = k2p[j0 + 16 + (lane & 15)];

        // ---- online softmax over the 32 new scores per row ----
#pragma unroll
        for (int r = 0; r < 8; ++r) {
            const float qk0 = c0[r], qk1 = c1[r];
            const float l0 = qk0 * INV_SQRT_DH - (q2r[r] + k2a - 2.0f * qk0) * wfr[r];
            const float l1 = qk1 * INV_SQRT_DH - (q2r[r] + k2b - 2.0f * qk1) * wfr[r];

            float mx = fmaxf(l0, l1);
            mx = fmaxf(mx, __shfl_xor(mx, 1, 32));
            mx = fmaxf(mx, __shfl_xor(mx, 2, 32));
            mx = fmaxf(mx, __shfl_xor(mx, 4, 32));
            mx = fmaxf(mx, __shfl_xor(mx, 8, 32));

            const float nm    = fmaxf(rmax[r], mx);
            const float scale = __expf(rmax[r] - nm);
            rmax[r] = nm;

            const float p0 = __expf(l0 - nm);
            const float p1 = __expf(l1 - nm);
            float ps = p0 + p1;
            ps += __shfl_xor(ps, 1, 32);
            ps += __shfl_xor(ps, 2, 32);
            ps += __shfl_xor(ps, 4, 32);
            ps += __shfl_xor(ps, 8, 32);
            rsum[r] = rsum[r] * scale + ps;

            acc[0][r] *= scale;
            acc[1][r] *= scale;
            acc[2][r] *= scale;
            acc[3][r] *= scale;

            const int m = ((lane >> 4) << 3) + r;
            Plds[wave][m][lane & 15]        = (_Float16)p0;
            Plds[wave][m][16 + (lane & 15)] = (_Float16)p1;
        }
        __syncthreads();

        // P in A-layout (16 x 32): two ds_load_b128
        const _Float16* pp = &Plds[wave][mrow][kbA];
        const v16h ap = ld2x8(pp, pp + 16);

        // ---- O += P V from LDS (B[k=key][n=d] = Vtl[d][key]) ----
#pragma unroll
        for (int nt = 0; nt < 4; ++nt) {
            const _Float16* vb = Vtl + (nt * 16 + (lane & 15)) * 32 + kbA;
            const v16h bv = ld2x8(vb, vb + 16);
            acc[nt] = WMMA_F16(ap, bv, acc[nt]);
        }
    }

    // ---- normalize and store to Oh[(b*N + q)*DIM + h*64 + d] ----
    const int b_ = bh >> 3, h = bh & 7;
#pragma unroll
    for (int r = 0; r < 8; ++r) {
        const int m = ((lane >> 4) << 3) + r;
        const float inv = 1.0f / rsum[r];
        const size_t base =
            ((size_t)(b_ * NN) + q0 + m) * DIM + h * DH + (lane & 15);
        Oh[base +  0] = (_Float16)(acc[0][r] * inv);
        Oh[base + 16] = (_Float16)(acc[1][r] * inv);
        Oh[base + 32] = (_Float16)(acc[2][r] * inv);
        Oh[base + 48] = (_Float16)(acc[3][r] * inv);
    }
}

// ---------------------------------------------------------------------------
// Final projection: out[4096,512] = Oh x Wo^T(stored transposed) + bo  (fp32)
// ---------------------------------------------------------------------------
__global__ void gemm_out_kernel(const _Float16* __restrict__ Oh,
                                const _Float16* __restrict__ Wot,
                                const float* __restrict__ bo,
                                float* __restrict__ out) {
    const int lane = threadIdx.x & 31;
    const int wave = threadIdx.x >> 5;
    const int row0 = blockIdx.x * 16;
    const int col0 = blockIdx.y * 128 + wave * 16;

    const int rowA = row0 + (lane & 15);
    const int nB   = col0 + (lane & 15);
    const int kbA  = (lane >> 4) * 8;
    const int koB  = (lane >> 4) * 16;

    const _Float16* aRow = Oh  + (size_t)rowA * DIM;
    const _Float16* bRow = Wot + (size_t)nB * DIM;

    v8f c = {};
    for (int k0 = 0; k0 < DIM; k0 += 32) {
        const v16h a = ld2x8(aRow + k0 + kbA, aRow + k0 + kbA + 16);
        if (k0 + 32 < DIM)
            __builtin_prefetch(bRow + k0 + 32 + koB, 0, 1);
        const v16h b = ld2x8(bRow + k0 + kbA, bRow + k0 + kbA + 16);
        c = WMMA_F16(a, b, c);
    }

    const float bias = bo[nB];
#pragma unroll
    for (int r = 0; r < 8; ++r) {
        const int m = ((lane >> 4) << 3) + r;
        out[(size_t)(row0 + m) * DIM + nB] = c[r] + bias;
    }
}

// ---------------------------------------------------------------------------
// Launch
// ---------------------------------------------------------------------------
extern "C" void kernel_launch(void* const* d_in, const int* in_sizes, int n_in,
                              void* d_out, int out_size, void* d_ws, size_t ws_size,
                              hipStream_t stream) {
    const float* z  = (const float*)d_in[0];
    const float* Wq = (const float*)d_in[1];
    const float* bq = (const float*)d_in[2];
    const float* Wk = (const float*)d_in[3];
    const float* bk = (const float*)d_in[4];
    const float* Wv = (const float*)d_in[5];
    const float* bv = (const float*)d_in[6];
    const float* Wo = (const float*)d_in[7];
    const float* bo = (const float*)d_in[8];

    char* w = (char*)d_ws;
    auto carve = [&](size_t bytes) {
        void* p = (void*)w;
        w += (bytes + 255) & ~(size_t)255;
        return p;
    };
    _Float16* zh  = (_Float16*)carve((size_t)MTOT * DIM * 2);
    _Float16* Wqt = (_Float16*)carve((size_t)DIM * DIM * 2);
    _Float16* Wkt = (_Float16*)carve((size_t)DIM * DIM * 2);
    _Float16* Wvt = (_Float16*)carve((size_t)DIM * DIM * 2);
    _Float16* Wot = (_Float16*)carve((size_t)DIM * DIM * 2);
    _Float16* Qh  = (_Float16*)carve((size_t)MTOT * DIM * 2);
    _Float16* Kh  = (_Float16*)carve((size_t)MTOT * DIM * 2);
    _Float16* Vt  = (_Float16*)carve((size_t)MTOT * DIM * 2);
    _Float16* Oh  = (_Float16*)carve((size_t)MTOT * DIM * 2);
    float* q2 = (float*)carve((size_t)BB * HH * NN * 4);
    float* k2 = (float*)carve((size_t)BB * HH * NN * 4);
    float* wf = (float*)carve((size_t)BB * HH * NN * 4);

    const int nz = MTOT * DIM;
    f32_to_f16_kernel<<<(nz + 255) / 256, 256, 0, stream>>>(z, zh, nz);

    dim3 gT(DIM / 32, DIM / 32);       // 16 x 16
    transpose_w_kernel<<<gT, 256, 0, stream>>>(Wq, Wqt);
    transpose_w_kernel<<<gT, 256, 0, stream>>>(Wk, Wkt);
    transpose_w_kernel<<<gT, 256, 0, stream>>>(Wv, Wvt);
    transpose_w_kernel<<<gT, 256, 0, stream>>>(Wo, Wot);

    dim3 gGemm(MTOT / 16, DIM / 128);  // 256 x 4
    gemm_qkv_kernel<<<gGemm, 256, 0, stream>>>(zh, Wqt, Wkt, Wvt, bq, bk, bv,
                                               Qh, Kh, Vt);

    row_stats_kernel<<<(BB * HH * NN) / 8, 256, 0, stream>>>(Qh, Kh, q2, k2, wf);

    attention_kernel<<<(BB * HH * NN / 16) / 8, 256, 0, stream>>>(
        Qh, Kh, Vt, q2, k2, wf, Oh);

    gemm_out_kernel<<<gGemm, 256, 0, stream>>>(Oh, Wot, bo, (float*)d_out);
}
